// RGCN_22333829939343
// MI455X (gfx1250) — compile-verified, matrix operand order checked by reading, not run on hardware
//
#include <hip/hip_runtime.h>
#include <hip/hip_bf16.h>
#include <stdint.h>

// ---- CDNA5 (gfx1250) RGCN: CSR gather + relation-fused bf16 WMMA GEMM ----

typedef __bf16 bf16_t;
typedef __attribute__((ext_vector_type(16))) __bf16 v16bf;
typedef __attribute__((ext_vector_type(8)))  __bf16 v8bf;
typedef __attribute__((ext_vector_type(4)))  __bf16 v4bf;
typedef __attribute__((ext_vector_type(8)))  float  v8f;

#define FEAT 128
#define NREL 3
#define OUTF 64
#define NEG_SLOPE 0.01f

// ---------------- utility kernels ----------------

__global__ void k_zero_i32(int* __restrict__ p, long n) {
    long i = (long)blockIdx.x * blockDim.x + threadIdx.x;
    if (i < n) p[i] = 0;
}

__global__ void k_zero_f32(float* __restrict__ p, long n) {
    long i = (long)blockIdx.x * blockDim.x + threadIdx.x;
    if (i < n) p[i] = 0.0f;
}

// float degree counts for both endpoints (for rsqrt norms)
__global__ void k_count_deg(const int* __restrict__ src, const int* __restrict__ dst,
                            float* __restrict__ odeg, float* __restrict__ ideg, int nE) {
    int e = blockIdx.x * blockDim.x + threadIdx.x;
    if (e >= nE) return;
    atomicAdd(&odeg[src[e]], 1.0f);
    atomicAdd(&ideg[dst[e]], 1.0f);
}

// int in-degree counts (CSR histogram)
__global__ void k_count_int(const int* __restrict__ dst, int* __restrict__ cnt, int nE) {
    int e = blockIdx.x * blockDim.x + threadIdx.x;
    if (e >= nE) return;
    atomicAdd(&cnt[dst[e]], 1);
}

__global__ void k_rsqrt_inplace(float* __restrict__ d, long n) {
    long i = (long)blockIdx.x * blockDim.x + threadIdx.x;
    if (i < n) d[i] = rsqrtf(fmaxf(d[i], 1.0f));
}

// single-block exclusive scan: off[0..n] from cnt[0..n-1]
__global__ void k_scan_excl(const int* __restrict__ cnt, int* __restrict__ off, int n) {
    __shared__ int carry;
    __shared__ int tmp[1024];
    if (threadIdx.x == 0) carry = 0;
    __syncthreads();
    for (int base = 0; base < n; base += (int)blockDim.x) {
        int i = base + (int)threadIdx.x;
        int v = (i < n) ? cnt[i] : 0;
        tmp[threadIdx.x] = v;
        __syncthreads();
        for (int s = 1; s < (int)blockDim.x; s <<= 1) {
            int t = (threadIdx.x >= (unsigned)s) ? tmp[threadIdx.x - s] : 0;
            __syncthreads();
            tmp[threadIdx.x] += t;
            __syncthreads();
        }
        int incl = tmp[threadIdx.x];
        if (i < n) off[i] = carry + incl - v;     // exclusive prefix
        __syncthreads();
        if (threadIdx.x == blockDim.x - 1) carry += incl;
        __syncthreads();
    }
    if (threadIdx.x == 0) off[n] = carry;
}

// counting-sort fill: csr_src grouped by destination node
__global__ void k_csr_fill(const int* __restrict__ src, const int* __restrict__ dst,
                           const int* __restrict__ off, int* __restrict__ cursor,
                           int* __restrict__ csr_src, int nE) {
    int e = blockIdx.x * blockDim.x + threadIdx.x;
    if (e >= nE) return;
    int d = dst[e];
    int p = atomicAdd(&cursor[d], 1);
    csr_src[off[d] + p] = src[e];
}

// W[din][dout] (row-major) -> Wt[dout][din] bf16 (K contiguous per output column)
__global__ void k_wconv(const float* __restrict__ W, bf16_t* __restrict__ Wt,
                        int din, int dout) {
    int idx = blockIdx.x * blockDim.x + threadIdx.x;
    if (idx >= din * dout) return;
    int o = idx / din, i = idx - o * din;
    Wt[(long)o * din + i] = (bf16_t)W[(long)i * dout + o];
}

// plain f32 -> bf16 convert
__global__ void k_to_bf16(const float* __restrict__ a, bf16_t* __restrict__ o, long total) {
    long i = (long)blockIdx.x * blockDim.x + threadIdx.x;
    if (i < total) o[i] = (bf16_t)a[i];
}

// ---------------- CSR gather aggregation (no atomics), all relations ----------------
// one wave per (relation, destination node); lane handles 4 features; 2-way edge unroll.
// aggb[r][d][:] = bf16( norm_dst[r][d] * sum_e norm_src[r][src_e] * h[src_e][:] )
__global__ void k_gather(const float* __restrict__ h, const int* __restrict__ csr_src,
                         const int* __restrict__ csr_off, const float* __restrict__ norm_src,
                         const float* __restrict__ norm_dst, bf16_t* __restrict__ aggb,
                         int N, int E) {
    int lane = threadIdx.x & 31;
    long gw = ((long)blockIdx.x * blockDim.x + threadIdx.x) >> 5;
    if (gw >= (long)NREL * N) return;
    int r = (int)(gw / N);
    int d = (int)(gw - (long)r * N);
    const int*   srcs = csr_src + (long)r * E;
    const int*   off  = csr_off + (long)r * (N + 1);
    const float* ns   = norm_src + (long)r * N;

    int beg = off[d], end = off[d + 1];
    float4 acc = make_float4(0.f, 0.f, 0.f, 0.f);
    int e = beg;
    for (; e + 1 < end; e += 2) {
        int s0 = srcs[e], s1 = srcs[e + 1];              // wave-uniform
        float n0 = ns[s0], n1 = ns[s1];
        float4 v0 = reinterpret_cast<const float4*>(h + (long)s0 * FEAT)[lane];
        float4 v1 = reinterpret_cast<const float4*>(h + (long)s1 * FEAT)[lane];
        acc.x += v0.x * n0 + v1.x * n1;
        acc.y += v0.y * n0 + v1.y * n1;
        acc.z += v0.z * n0 + v1.z * n1;
        acc.w += v0.w * n0 + v1.w * n1;
    }
    if (e < end) {
        int s0 = srcs[e];
        float n0 = ns[s0];
        float4 v0 = reinterpret_cast<const float4*>(h + (long)s0 * FEAT)[lane];
        acc.x += v0.x * n0; acc.y += v0.y * n0;
        acc.z += v0.z * n0; acc.w += v0.w * n0;
    }
    float nd = norm_dst[(long)r * N + d];
    v4bf o;
    o[0] = (bf16_t)(acc.x * nd);
    o[1] = (bf16_t)(acc.y * nd);
    o[2] = (bf16_t)(acc.z * nd);
    o[3] = (bf16_t)(acc.w * nd);
    *reinterpret_cast<v4bf*>(aggb + (gw * FEAT) + lane * 4) = o;
}

// ---------------- relation-fused WMMA GEMM ----------------
// C[n][col] = act( scale * ( sum_r A_r[n][:] @ B_r[col][:] + sum_r bias_r[col] ) )
// A: bf16 [nrel][nrows][128]; B: bf16 [nrel][ncols][128] (pre-transposed weights)
// one wave computes one 16x16 output tile; nrel*4 x v_wmma_f32_16x16x32_bf16
// REQUIRES nrows % 16 == 0 (true here: 50000 = 3125*16)
__global__ void k_wmma_gemm(const bf16_t* __restrict__ A, long aStride,
                            const bf16_t* __restrict__ B, long bStride,
                            const float* __restrict__ bias, long biasStride, int nrel,
                            float scale, int do_relu,
                            float* __restrict__ C, int nrows, int nct, int ldc) {
    int wave = (int)(((long)blockIdx.x * blockDim.x + threadIdx.x) >> 5);
    int nrt = nrows >> 4;
    if (wave >= nrt * nct) return;           // wave-uniform: EXEC all-ones for WMMA
    int rt = wave / nct;
    int ct = wave - rt * nct;
    int lane = threadIdx.x & 31;
    int m = lane & 15;
    int half = lane >> 4;

    v8f acc = {};
    for (int r = 0; r < nrel; ++r) {
        const bf16_t* ap = A + r * aStride + (long)(rt * 16 + m) * FEAT;
        const bf16_t* bp = B + r * bStride + (long)(ct * 16 + m) * FEAT;
#pragma unroll
        for (int kk = 0; kk < 4; ++kk) {
            int kb = kk * 32 + half * 8;
            union { v16bf v; v8bf h[2]; } ua, ub;
            ua.h[0] = *reinterpret_cast<const v8bf*>(ap + kb);        // K = kb..kb+7
            ua.h[1] = *reinterpret_cast<const v8bf*>(ap + kb + 16);   // K = kb+16..kb+23
            ub.h[0] = *reinterpret_cast<const v8bf*>(bp + kb);
            ub.h[1] = *reinterpret_cast<const v8bf*>(bp + kb + 16);
            acc = __builtin_amdgcn_wmma_f32_16x16x32_bf16(
                false, ua.v, false, ub.v, (short)0, acc, false, false);
        }
    }

    int col = ct * 16 + m;
    float bv = 0.0f;
    for (int r = 0; r < nrel; ++r) bv += bias[r * biasStride + col];

    float* cp = C + (long)(rt * 16 + half * 8) * ldc + col;
#pragma unroll
    for (int v = 0; v < 8; ++v) {
        float val = (acc[v] + bv) * scale;
        if (do_relu) val = (val > 0.0f) ? val : NEG_SLOPE * val;
        cp[(long)v * ldc] = val;
    }
}

// ---------------- host orchestration ----------------

static inline int div_up(long a, long b) { return (int)((a + b - 1) / b); }

extern "C" void kernel_launch(void* const* d_in, const int* in_sizes, int n_in,
                              void* d_out, int out_size, void* d_ws, size_t ws_size,
                              hipStream_t stream) {
    (void)n_in; (void)out_size; (void)ws_size;
    const float* x    = (const float*)d_in[0];
    const int*   esrc = (const int*)d_in[1];
    const int*   edst = (const int*)d_in[2];
    const float* W0   = (const float*)d_in[3];
    const float* b0   = (const float*)d_in[4];
    const float* Wl   = (const float*)d_in[5];
    const float* bl   = (const float*)d_in[6];
    const float* Wout = (const float*)d_in[7];
    const float* bout = (const float*)d_in[8];
    float* out = (float*)d_out;

    const int N = in_sizes[0] / FEAT;     // 50000
    const int E = in_sizes[1] / NREL;     // 800000

    // workspace carve-up (256B aligned)
    char* w = (char*)d_ws;
    auto carve = [&](size_t bytes) -> void* {
        void* p = (void*)w;
        w += (bytes + 255) & ~(size_t)255;
        return p;
    };
    float*  hA      = (float*)carve((size_t)N * FEAT * sizeof(float));
    float*  hB      = (float*)carve((size_t)N * FEAT * sizeof(float));
    bf16_t* aggb    = (bf16_t*)carve((size_t)NREL * N * FEAT * sizeof(bf16_t));
    float*  norms   = (float*)carve((size_t)2 * NREL * N * sizeof(float));
    int*    csr_src = (int*)carve((size_t)NREL * E * sizeof(int));
    int*    csr_off = (int*)carve((size_t)NREL * (N + 1) * sizeof(int));
    int*    scratch = (int*)carve((size_t)NREL * N * sizeof(int));   // counts, then cursor
    bf16_t* WtAll   = (bf16_t*)carve((size_t)5 * NREL * FEAT * FEAT * sizeof(bf16_t));
    bf16_t* Wtout   = (bf16_t*)carve((size_t)FEAT * OUTF * sizeof(bf16_t));
    float* norm_src = norms;
    float* norm_dst = norms + (size_t)NREL * N;

    const int TB = 256;

    // 1) degree -> rsqrt norms (edges fixed across layers: compute once)
    {
        long n = (long)2 * NREL * N;
        k_zero_f32<<<div_up(n, TB), TB, 0, stream>>>(norms, n);
        for (int r = 0; r < NREL; ++r)
            k_count_deg<<<div_up(E, TB), TB, 0, stream>>>(
                esrc + (long)r * E, edst + (long)r * E,
                norm_src + (long)r * N, norm_dst + (long)r * N, E);
        k_rsqrt_inplace<<<div_up(n, TB), TB, 0, stream>>>(norms, n);
    }

    // 2) CSR-by-destination per relation (count -> scan -> fill), built once
    {
        long n = (long)NREL * N;
        k_zero_i32<<<div_up(n, TB), TB, 0, stream>>>(scratch, n);
        for (int r = 0; r < NREL; ++r)
            k_count_int<<<div_up(E, TB), TB, 0, stream>>>(
                edst + (long)r * E, scratch + (long)r * N, E);
        for (int r = 0; r < NREL; ++r)
            k_scan_excl<<<1, 1024, 0, stream>>>(
                scratch + (long)r * N, csr_off + (long)r * (N + 1), N);
        k_zero_i32<<<div_up(n, TB), TB, 0, stream>>>(scratch, n);    // reuse as cursor
        for (int r = 0; r < NREL; ++r)
            k_csr_fill<<<div_up(E, TB), TB, 0, stream>>>(
                esrc + (long)r * E, edst + (long)r * E,
                csr_off + (long)r * (N + 1), scratch + (long)r * N,
                csr_src + (long)r * E, E);
    }

    // 3) weights -> transposed bf16 [out][in]
    {
        const int wN = FEAT * FEAT;
        for (int r = 0; r < NREL; ++r)
            k_wconv<<<div_up(wN, TB), TB, 0, stream>>>(
                W0 + (long)r * wN, WtAll + (long)r * wN, FEAT, FEAT);
        for (int l = 0; l < 4; ++l)
            for (int r = 0; r < NREL; ++r)
                k_wconv<<<div_up(wN, TB), TB, 0, stream>>>(
                    Wl + ((long)l * NREL + r) * wN,
                    WtAll + ((long)(l + 1) * NREL + r) * wN, FEAT, FEAT);
        k_wconv<<<div_up(FEAT * OUTF, TB), TB, 0, stream>>>(Wout, Wtout, FEAT, OUTF);
    }

    // 4) five hetero-conv layers: one gather launch + one relation-fused WMMA GEMM
    const long tot      = (long)N * FEAT;
    const long aStride  = (long)N * FEAT;            // bf16 elements per relation
    const long bStride  = (long)FEAT * FEAT;
    const int  nct      = FEAT / 16;                 // 8 column tiles
    const int  nrt      = N / 16;                    // 3125 row tiles (exact)
    const int  gemmBl   = div_up((long)nrt * nct * 32, TB);
    const int  gatherBl = div_up((long)NREL * N * 32, TB);

    const float* hin = x;
    float* hout = hA;
    for (int l = 0; l < 5; ++l) {
        const bf16_t* Wt   = WtAll + (size_t)l * NREL * FEAT * FEAT;
        const float*  bias = (l == 0) ? b0 : (bl + (size_t)(l - 1) * NREL * FEAT);
        k_gather<<<gatherBl, TB, 0, stream>>>(
            hin, csr_src, csr_off, norm_src, norm_dst, aggb, N, E);
        k_wmma_gemm<<<gemmBl, TB, 0, stream>>>(
            aggb, aStride, Wt, bStride, bias, (long)FEAT, NREL,
            1.0f / (float)NREL, (l != 4) ? 1 : 0,
            hout, N, nct, FEAT);
        hin = hout;
        hout = (hout == hA) ? hB : hA;
    }

    // 5) final linear 128 -> 64 (same kernel, nrel=1, no scale/relu)
    k_to_bf16<<<div_up(tot, TB), TB, 0, stream>>>(hin, aggb, tot);
    {
        const int fct = OUTF / 16;                   // 4 column tiles
        const int fbl = div_up((long)nrt * fct * 32, TB);
        k_wmma_gemm<<<fbl, TB, 0, stream>>>(
            aggb, aStride, Wtout, bStride, bout, (long)OUTF, 1,
            1.0f, 0, out, N, fct, OUTF);
    }
}
